// TAA_Spectral_67619965108635
// MI455X (gfx1250) — matrix operand order repaired and checked by reading, not compiled, not run on hardware
//
#include <hip/hip_runtime.h>

// ---------------------------------------------------------------------------
// Problem constants (from reference): N=65536 tokens, E=256, 8 heads, hd=32,
// chunk=256 -> 256 independent attention blocks.
// ---------------------------------------------------------------------------
#define EDIM      256
#define NTOK      65536
#define NCHUNK    256
#define NHEAD     8
#define HDIM      32
#define CHUNKROWS 256

typedef __attribute__((ext_vector_type(16))) _Float16 v16h;
typedef __attribute__((ext_vector_type(8)))  _Float16 v8h;
typedef __attribute__((ext_vector_type(8)))  float    v8f;
typedef __attribute__((ext_vector_type(4)))  float    v4f;

static __device__ __forceinline__ v8f vzero8() {
  v8f z = {0.f, 0.f, 0.f, 0.f, 0.f, 0.f, 0.f, 0.f};
  return z;
}

static __device__ __forceinline__ v16h pack16(v8h lo, v8h hi) {
  v16h r;
#pragma unroll
  for (int i = 0; i < 8; ++i) { r[i] = lo[i]; r[8 + i] = hi[i]; }
  return r;
}

// A operand (16x32 f16 tile, M x K). Source row-major [row][k], K contiguous.
// ISA layout: lane m (0-15): K={0..7,16..23}; lane m+16: K={8..15,24..31}.
static __device__ __forceinline__ v16h load_a(const _Float16* s, int stride,
                                              int r0, int k0, int lane) {
  int m = lane & 15, h = (lane >> 4) & 1;
  const _Float16* p = s + (size_t)(r0 + m) * stride + k0 + h * 8;
  return pack16(*(const v8h*)p, *(const v8h*)(p + 16));
}

// B operand (32x16 f16 tile, K x N). Source stored as [col][k], K contiguous
// (i.e. transposed / "NT" weight layout). Lane n (0-15): K={0..7,16..23}; lane n+16:
// K={8..15,24..31}, column = lane&15.
static __device__ __forceinline__ v16h load_b(const _Float16* s, int stride,
                                              int n0, int k0, int lane) {
  int n = lane & 15, h = (lane >> 4) & 1;
  const _Float16* p = s + (size_t)(n0 + n) * stride + k0 + h * 8;
  return pack16(*(const v8h*)p, *(const v8h*)(p + 16));
}

static __device__ __forceinline__ v8f wmma16(v16h a, v16h b, v8f c) {
  return __builtin_amdgcn_wmma_f32_16x16x32_f16(false, a, false, b, (short)0, c,
                                                false, false);
}

// ---------------------------------------------------------------------------
// CDNA5 async global->LDS copy (16B per lane, tracked by ASYNCcnt).
// LDS dest address = low 32 bits of generic pointer.
// ---------------------------------------------------------------------------
static __device__ __forceinline__ void async_copy16(void* lds_dst,
                                                    const void* gsrc) {
  unsigned int loff = (unsigned int)(unsigned long long)lds_dst;
  unsigned long long ga = (unsigned long long)gsrc;
  asm volatile("global_load_async_to_lds_b128 %0, %1, off"
               :
               : "v"(loff), "v"(ga)
               : "memory");
}
static __device__ __forceinline__ void async_wait0() {
  asm volatile("s_wait_asynccnt 0x0" ::: "memory");
}

// ---------------------------------------------------------------------------
// Stage 0a: fuse per-stream projection with MHA in-projection.
//   Wf_p[o][j] = sum_i inproj[p*E+o][i] * base_p[i][j]      (f16 output)
// grid (E, 3), block E.  Bases are L2-resident (768 KB total vs 192 MB L2).
// ---------------------------------------------------------------------------
__global__ void taa_fuse_w(const float* __restrict__ b0,
                           const float* __restrict__ b1,
                           const float* __restrict__ b2,
                           const float* __restrict__ inproj,
                           _Float16* __restrict__ wf) {
  int p = blockIdx.y;
  const float* base = (p == 0) ? b0 : ((p == 1) ? b1 : b2);
  int o = blockIdx.x;
  int j = threadIdx.x;
  const float* wrow = inproj + ((size_t)p * EDIM + o) * EDIM;
  float acc = 0.f;
  for (int i = 0; i < EDIM; ++i) acc += wrow[i] * base[(size_t)i * EDIM + j];
  wf[((size_t)p * EDIM + o) * EDIM + j] = (_Float16)acc;
}

// Stage 0b: fused bias:  bf_p[o] = sum_i inproj[p*E+o][i]*base_b_p[i] + inprojb[p*E+o]
__global__ void taa_fuse_b(const float* __restrict__ hb,
                           const float* __restrict__ sb,
                           const float* __restrict__ cb,
                           const float* __restrict__ inproj,
                           const float* __restrict__ inprojb,
                           float* __restrict__ bf) {
  int p = blockIdx.x;
  int o = threadIdx.x;
  const float* bb = (p == 0) ? hb : ((p == 1) ? sb : cb);
  const float* wrow = inproj + ((size_t)p * EDIM + o) * EDIM;
  float acc = inprojb[p * EDIM + o];
  for (int i = 0; i < EDIM; ++i) acc += wrow[i] * bb[i];
  bf[p * EDIM + o] = acc;
}

// Stage 0c: out_w f32 -> f16 copy.
__global__ void taa_cvt_outw(const float* __restrict__ w,
                             _Float16* __restrict__ wh) {
  int i = blockIdx.x * blockDim.x + threadIdx.x;
  wh[i] = (_Float16)w[i];
}

// ---------------------------------------------------------------------------
// Stage 1: QKV projection.  Q[n][o] = sum_j X[n][j]*Wf[o][j] + bf[o]
// One workgroup per (chunk c, stream p in {q,k,v}); 256 threads = 8 waves.
// Wf staged in LDS via async-to-LDS copies (padded stride 264 halves ->
// conflict-free WMMA B reads); A hoisted from global f32 with on-the-fly f16
// convert.  Output layout head-major: qkv[p][(c*8+h)*256 + r][d]  (f16).
// ---------------------------------------------------------------------------
__global__ void __launch_bounds__(256)
taa_qkv_proj(const float* __restrict__ x0, const float* __restrict__ x1,
             const float* __restrict__ x2, const _Float16* __restrict__ wf,
             const float* __restrict__ bf, _Float16* __restrict__ qkv) {
  extern __shared__ _Float16 sW[];  // [256][264]
  const int c = blockIdx.x, p = blockIdx.y;
  const float* X = ((p == 0) ? x0 : ((p == 1) ? x1 : x2)) +
                   (size_t)c * CHUNKROWS * EDIM;
  const _Float16* W = wf + (size_t)p * EDIM * EDIM;
  const float* bias = bf + p * EDIM;
  _Float16* out = qkv + (size_t)p * NTOK * EDIM;
  const int t = threadIdx.x;

  // stage Wf -> LDS: 8192 x 16B async copies, coalesced
  for (int k = 0; k < 32; ++k) {
    int idx = t + k * 256;
    int row = idx >> 5, c16 = idx & 31;
    async_copy16(&sW[row * 264 + c16 * 8], &W[(size_t)row * EDIM + c16 * 8]);
  }
  async_wait0();
  __syncthreads();

  const int w = t >> 5, lane = t & 31;
  const int m = lane & 15, hh = (lane >> 4) & 1;
  for (int rt2 = 0; rt2 < 2; ++rt2) {
    int rt = w * 2 + rt2;  // 16-row tile index, rows rt*16..rt*16+15
    // hoist A operand: 8 K-tiles of this row tile, f32->f16 convert
    v16h a[8];
    const float* xrow = X + (size_t)(rt * 16 + m) * EDIM;
#pragma unroll
    for (int kt = 0; kt < 8; ++kt) {
      const float* pk = xrow + kt * 32 + hh * 8;
      v4f f0 = *(const v4f*)(pk);
      v4f f1 = *(const v4f*)(pk + 4);
      v4f f2 = *(const v4f*)(pk + 16);
      v4f f3 = *(const v4f*)(pk + 20);
      v16h av;
#pragma unroll
      for (int i = 0; i < 4; ++i) {
        av[i]      = (_Float16)f0[i];
        av[4 + i]  = (_Float16)f1[i];
        av[8 + i]  = (_Float16)f2[i];
        av[12 + i] = (_Float16)f3[i];
      }
      a[kt] = av;
    }
    for (int ct = 0; ct < 16; ++ct) {
      v8f acc = vzero8();
#pragma unroll
      for (int kt = 0; kt < 8; ++kt) {
        v16h b = load_b(sW, 264, ct * 16, kt * 32, lane);
        acc = wmma16(a[kt], b, acc);
      }
      int o = ct * 16 + m;
      float bo = bias[o];
      int h = o >> 5, d = o & 31;  // head, dim-in-head
      size_t rowbase = (size_t)(c * NHEAD + h) * CHUNKROWS;
#pragma unroll
      for (int vi = 0; vi < 8; ++vi) {
        int r = rt * 16 + vi + 8 * hh;
        out[(rowbase + r) * HDIM + d] = (_Float16)(acc[vi] + bo);
      }
    }
  }
}

// ---------------------------------------------------------------------------
// Stage 2: block-diagonal attention, one workgroup per (chunk, head).
// Q/K staged with async-to-LDS raw copies (the 1/sqrt(hd) scale is folded
// into the softmax: exp(scale*(s-max)) ).  S = Q K^T (one WMMA per 16x16 tile
// since K=hd=32), in-register softmax with 16-lane shfl_xor reductions,
// normalized P -> per-wave LDS scratch (D-layout -> A-layout), ctx = P V.
// ---------------------------------------------------------------------------
__global__ void __launch_bounds__(256)
taa_attn(const _Float16* __restrict__ q, const _Float16* __restrict__ k,
         const _Float16* __restrict__ v, _Float16* __restrict__ ctx) {
  extern __shared__ _Float16 sm[];
  _Float16* sQ  = sm;            // [256][40]   (padded: 20 dwords/row)
  _Float16* sK  = sm + 10240;    // [256][40]
  _Float16* sVt = sm + 20480;    // [32][264]   V transposed: Vt[d][r]
  _Float16* sP  = sm + 28928;    // 8 waves x [16][264]

  const int c = blockIdx.x, h = blockIdx.y;
  const size_t base = (size_t)(c * NHEAD + h) * CHUNKROWS * HDIM;
  const int t = threadIdx.x;
  const float scale = 0.17677669529663689f;  // 1/sqrt(32)

  {  // stage Q, K via async copies; V^T via manual scatter
    const _Float16* qr = q + base + (size_t)t * HDIM;
    const _Float16* kr = k + base + (size_t)t * HDIM;
#pragma unroll
    for (int i = 0; i < 4; ++i) {
      async_copy16(&sQ[t * 40 + i * 8], qr + i * 8);
      async_copy16(&sK[t * 40 + i * 8], kr + i * 8);
    }
    const _Float16* vr = v + base + (size_t)t * HDIM;
#pragma unroll
    for (int d = 0; d < HDIM; ++d) sVt[d * 264 + t] = vr[d];
    async_wait0();
  }
  __syncthreads();

  const int w = t >> 5, lane = t & 31;
  const int m = lane & 15, hh = (lane >> 4) & 1;
  _Float16* sPw = sP + w * 16 * 264;

  for (int rt2 = 0; rt2 < 2; ++rt2) {
    int rt = w * 2 + rt2;
    v16h aq = load_a(sQ, 40, rt * 16, 0, lane);
    v8f s[16];
#pragma unroll
    for (int ct = 0; ct < 16; ++ct) {
      v16h b = load_b(sK, 40, ct * 16, 0, lane);
      s[ct] = wmma16(aq, b, vzero8());
    }
    // row softmax: row = rt*16 + vi + 8*hh; reduce over 16-lane half-wave
#pragma unroll
    for (int vi = 0; vi < 8; ++vi) {
      float mx = -3.0e38f;
#pragma unroll
      for (int ct = 0; ct < 16; ++ct) mx = fmaxf(mx, s[ct][vi]);
#pragma unroll
      for (int off = 8; off >= 1; off >>= 1)
        mx = fmaxf(mx, __shfl_xor(mx, off, 32));
      float sum = 0.f;
#pragma unroll
      for (int ct = 0; ct < 16; ++ct) {
        float e = __expf(scale * (s[ct][vi] - mx));  // scale folded in
        s[ct][vi] = e;
        sum += e;
      }
#pragma unroll
      for (int off = 8; off >= 1; off >>= 1) sum += __shfl_xor(sum, off, 32);
      float inv = 1.f / sum;
#pragma unroll
      for (int ct = 0; ct < 16; ++ct)
        sPw[(vi + 8 * hh) * 264 + ct * 16 + m] = (_Float16)(s[ct][vi] * inv);
    }
    __syncthreads();  // uniform across all 8 waves; fences P writes->reads

    // ctx tile (16 rows x 32 dims) = P[16x256] @ V[256x32]
    v8f c0 = vzero8(), c1 = vzero8();
#pragma unroll
    for (int kt = 0; kt < 8; ++kt) {
      v16h ap = load_a(sPw, 264, 0, kt * 32, lane);
      v16h b0 = load_b(sVt, 264, 0, kt * 32, lane);
      v16h b1 = load_b(sVt, 264, 16, kt * 32, lane);
      c0 = wmma16(ap, b0, c0);
      c1 = wmma16(ap, b1, c1);
    }
    _Float16* op = ctx + base;
#pragma unroll
    for (int vi = 0; vi < 8; ++vi) {
      int r = rt * 16 + vi + 8 * hh;
      op[(size_t)r * HDIM + m]      = (_Float16)c0[vi];
      op[(size_t)r * HDIM + 16 + m] = (_Float16)c1[vi];
    }
    __syncthreads();  // WAR fence before next row-tile reuses sPw
  }
}

// ---------------------------------------------------------------------------
// Stage 3: output projection.  out[n][o] = sum_e ctx[n][e]*out_w[o][e]+out_b[o]
// ctx staged in LDS via async copies (head-gather into row-major [256][264]);
// out_w (f16) streamed through L2 (128 KB, shared by all 256 blocks).
// ---------------------------------------------------------------------------
__global__ void __launch_bounds__(256)
taa_out_proj(const _Float16* __restrict__ ctx, const _Float16* __restrict__ ow,
             const float* __restrict__ ob, float* __restrict__ out) {
  extern __shared__ _Float16 sA[];  // [256][264]
  const int c = blockIdx.x;
  const int t = threadIdx.x;

  // stage ctx chunk: gather head-major [c][h][r][d] -> row-major [r][h*32+d]
  for (int kk = 0; kk < 32; ++kk) {
    int idx = t + kk * 256;              // 8192 x 16B chunks
    int h = idx >> 10;                   // 1024 chunks per head
    int rem = idx & 1023;
    int r = rem >> 2, c4 = rem & 3;
    const _Float16* src =
        ctx + ((size_t)(c * NHEAD + h) * CHUNKROWS + r) * HDIM + c4 * 8;
    async_copy16(&sA[r * 264 + h * 32 + c4 * 8], src);
  }
  async_wait0();
  __syncthreads();

  const int w = t >> 5, lane = t & 31;
  const int m = lane & 15, hh = (lane >> 4) & 1;
  for (int rt2 = 0; rt2 < 2; ++rt2) {
    int rt = w * 2 + rt2;
    v16h a[8];
#pragma unroll
    for (int kt = 0; kt < 8; ++kt) a[kt] = load_a(sA, 264, rt * 16, kt * 32, lane);
    for (int ct = 0; ct < 16; ++ct) {
      v8f acc = vzero8();
#pragma unroll
      for (int kt = 0; kt < 8; ++kt) {
        v16h b = load_b(ow, EDIM, ct * 16, kt * 32, lane);  // global, L2-hot
        acc = wmma16(a[kt], b, acc);
      }
      int o = ct * 16 + m;
      float bo = ob[o];
#pragma unroll
      for (int vi = 0; vi < 8; ++vi) {
        int r = c * CHUNKROWS + rt * 16 + vi + 8 * hh;
        out[(size_t)r * EDIM + o] = acc[vi] + bo;
      }
    }
  }
}

// ---------------------------------------------------------------------------
// Host launcher.  Input order (setup_inputs):
//  0 X_L_hyp  1 X_L_star  2 X_L_clique  3 hyp_w 4 hyp_b 5 star_w 6 star_b
//  7 clique_w 8 clique_b  9 in_proj_w  10 in_proj_b 11 out_w 12 out_b
// 13 chunk_size (==256, hardcoded)
// ---------------------------------------------------------------------------
extern "C" void kernel_launch(void* const* d_in, const int* in_sizes, int n_in,
                              void* d_out, int out_size, void* d_ws,
                              size_t ws_size, hipStream_t stream) {
  const float* X0 = (const float*)d_in[0];
  const float* X1 = (const float*)d_in[1];
  const float* X2 = (const float*)d_in[2];
  const float* hyp_w = (const float*)d_in[3];
  const float* hyp_b = (const float*)d_in[4];
  const float* star_w = (const float*)d_in[5];
  const float* star_b = (const float*)d_in[6];
  const float* clique_w = (const float*)d_in[7];
  const float* clique_b = (const float*)d_in[8];
  const float* in_proj_w = (const float*)d_in[9];
  const float* in_proj_b = (const float*)d_in[10];
  const float* out_w = (const float*)d_in[11];
  const float* out_b = (const float*)d_in[12];
  float* outp = (float*)d_out;

  // workspace carve-up (all offsets 1 KB aligned)
  char* ws = (char*)d_ws;
  const size_t NE2 = (size_t)NTOK * EDIM * sizeof(_Float16);  // 32 MB
  _Float16* wsWf   = (_Float16*)(ws);                         // 384 KB
  float*    wsBias = (float*)(ws + 393216);                   // 3 KB
  _Float16* wsOutW = (_Float16*)(ws + 396288);                // 128 KB
  _Float16* wsQKV  = (_Float16*)(ws + 527360);                // 96 MB
  _Float16* wsCtx  = (_Float16*)(ws + 527360 + 3 * NE2);      // 32 MB

  // stage 0: weight fusion / conversion (tiny, L2-resident)
  taa_fuse_w<<<dim3(EDIM, 3), EDIM, 0, stream>>>(hyp_w, star_w, clique_w,
                                                 in_proj_w, wsWf);
  taa_fuse_b<<<3, EDIM, 0, stream>>>(hyp_b, star_b, clique_b, in_proj_w,
                                     in_proj_b, wsBias);
  taa_cvt_outw<<<EDIM, EDIM, 0, stream>>>(out_w, wsOutW);

  // stage 1: fused QKV projection (WMMA)
  taa_qkv_proj<<<dim3(NCHUNK, 3), 256, 256 * 264 * sizeof(_Float16), stream>>>(
      X0, X1, X2, wsWf, wsBias, wsQKV);

  // stage 2: chunked attention (WMMA + softmax)
  const size_t NE = (size_t)NTOK * EDIM;
  taa_attn<<<dim3(NCHUNK, NHEAD), 256, 62720 * sizeof(_Float16), stream>>>(
      wsQKV, wsQKV + NE, wsQKV + 2 * NE, wsCtx);

  // stage 3: output projection (WMMA)
  taa_out_proj<<<NCHUNK, 256, 256 * 264 * sizeof(_Float16), stream>>>(
      wsCtx, wsOutW, out_b, outp);
}
